// lstm_unidir_bsany_14894946582773
// MI455X (gfx1250) — compile-verified, matrix operand order checked by reading, not run on hardware
//
#include <hip/hip_runtime.h>
#include <hip/hip_bf16.h>

// ---------------- problem constants ----------------
#define TT   512
#define BB   64
#define IN_D 1024
#define HH   1024
#define LL   2
#define KK   2048          // IN+H == H+H
#define N4   4096          // 4*H
#define NTILES 256         // N4/16
#define KTILES 64          // KK/32

typedef __attribute__((ext_vector_type(16))) __bf16 v16bf;
typedef __attribute__((ext_vector_type(8)))  float  v8f;

__device__ __forceinline__ unsigned short f32_to_bf16_rne(float f) {
    union { float f; unsigned int u; } v; v.f = f;
    unsigned int lsb = (v.u >> 16) & 1u;
    unsigned int r = v.u + 0x7FFFu + lsb;
    return (unsigned short)(r >> 16);
}

__device__ __forceinline__ float fast_sigmoid(float x) {
    return 1.0f / (1.0f + __expf(-x));
}
__device__ __forceinline__ float fast_tanh(float x) {
    float e = __expf(2.0f * x);
    return (e - 1.0f) / (e + 1.0f);
}

// ---------------------------------------------------------------------------
// Convert fp32 weight [N4 x KK] row-major into bf16 WMMA-tile-swizzled layout:
// offset = ((nT*KTILES + kt)*512 + lane*16 + kpos)
// lane(0..31): n = nT*16 + (lane&15), k-half = (lane>>4)*16, kpos = 0..15
// ---------------------------------------------------------------------------
__global__ __launch_bounds__(256) void wconvert_tiled_kernel(
        const float* __restrict__ W, unsigned short* __restrict__ Wt) {
    int idx  = blockIdx.x * 256 + threadIdx.x;      // 0 .. N4*KK-1 (2^23)
    int e    = idx & 511;
    int kt   = (idx >> 9) & (KTILES - 1);
    int nT   = idx >> 15;                           // 0..255
    int lane = e >> 4;
    int kpos = e & 15;
    int n = nT * 16 + (lane & 15);
    int k = kt * 32 + (lane >> 4) * 16 + kpos;
    Wt[idx] = f32_to_bf16_rne(W[(size_t)n * KK + k]);
}

// ---------------------------------------------------------------------------
// Pack a [BB x 1024] fp32 panel into one half of the bf16 activation panel
// XH [BB x KK]; dst is pre-offset by 0 (x half) or 1024 (h half).
// ---------------------------------------------------------------------------
__global__ __launch_bounds__(256) void pack_bf16_kernel(
        const float* __restrict__ src, unsigned short* __restrict__ dst) {
    int idx = blockIdx.x * 256 + threadIdx.x;       // 0 .. BB*1024-1
    int b   = idx >> 10;
    int col = idx & 1023;
    dst[(size_t)b * KK + col] = f32_to_bf16_rne(src[idx]);
}

// ---------------------------------------------------------------------------
// Gates GEMM:  gates[64 x 4096] = XH[64 x 2048](bf16) x Wt^T  (fp32 accum)
// grid = 64 WGs (64 output cols each), block = 256 threads = 8 waves.
// Wave w: rows 16*(w&3).. , tile-cols blockIdx*4 + (w>>2)*2 (2 tiles/wave).
// Depth-2 software pipeline: loads for K-step kt+2 issued while WMMAs of
// kt execute, so the s_wait before each WMMA tolerates ~2 K-steps of L2
// latency instead of forcing loadcnt==0.
// ---------------------------------------------------------------------------
__global__ __launch_bounds__(256) void gemm_gates_wmma_kernel(
        const unsigned short* __restrict__ XH,
        const unsigned short* __restrict__ Wt,
        float* __restrict__ gates) {
    const int lane = threadIdx.x & 31;
    const int wave = threadIdx.x >> 5;
    const int m0   = (wave & 3) * 16;
    const int nT0  = blockIdx.x * 4 + (wave >> 2) * 2;   // first of 2 tile-cols

    v8f acc0 = {}, acc1 = {};

    const unsigned short* aptr =
        XH + (size_t)(m0 + (lane & 15)) * KK + (lane >> 4) * 16;
    const unsigned short* bp0 = Wt + ((size_t)(nT0 + 0) * KTILES) * 512 + lane * 16;
    const unsigned short* bp1 = Wt + ((size_t)(nT0 + 1) * KTILES) * 512 + lane * 16;

    auto LD = [&](int kt, v16bf& a, v16bf& b0, v16bf& b1) {
        a  = *(const v16bf*)(aptr + (size_t)kt * 32);
        b0 = *(const v16bf*)(bp0 + (size_t)kt * 512);
        b1 = *(const v16bf*)(bp1 + (size_t)kt * 512);
    };

    v16bf a0, b00, b10;   // pipeline slot 0 (even kt)
    v16bf a1, b01, b11;   // pipeline slot 1 (odd kt)
    LD(0, a0, b00, b10);
    LD(1, a1, b01, b11);

    for (int kt = 0; kt < KTILES; kt += 2) {
        // WGP-scope prefetch of the weight panels ~16 K-steps ahead
        // (overreads stay inside d_ws; speculative prefetch is droppable).
        __builtin_prefetch(bp0 + (size_t)(kt + 16) * 512, 0, 3);
        __builtin_prefetch(bp1 + (size_t)(kt + 16) * 512, 0, 3);

        acc0 = __builtin_amdgcn_wmma_f32_16x16x32_bf16(false, a0, false, b00,
                                                       (short)0, acc0, false, false);
        acc1 = __builtin_amdgcn_wmma_f32_16x16x32_bf16(false, a0, false, b10,
                                                       (short)0, acc1, false, false);
        LD(kt + 2, a0, b00, b10);    // tail overreads (<=2 tiles) stay in d_ws

        acc0 = __builtin_amdgcn_wmma_f32_16x16x32_bf16(false, a1, false, b01,
                                                       (short)0, acc0, false, false);
        acc1 = __builtin_amdgcn_wmma_f32_16x16x32_bf16(false, a1, false, b11,
                                                       (short)0, acc1, false, false);
        LD(kt + 3, a1, b01, b11);
    }

    // C/D layout: VGPR r holds element (M = m0 + r + 8*(lane>>4), N = lane&15)
    const int row0 = m0 + (lane >> 4) * 8;
    const int col  = lane & 15;
    float* g0 = gates + (size_t)row0 * N4 + (nT0 + 0) * 16 + col;
    float* g1 = gates + (size_t)row0 * N4 + (nT0 + 1) * 16 + col;
#pragma unroll
    for (int r = 0; r < 8; ++r) {
        g0[(size_t)r * N4] = acc0[r];
        g1[(size_t)r * N4] = acc1[r];
    }
}

// ---------------------------------------------------------------------------
// Fused LSTM cell: bias add + gate activations + c/h update.
// Writes fp32 c & h state, bf16 h into up to two activation-panel halves,
// and optionally hlast[t].
// ---------------------------------------------------------------------------
__global__ __launch_bounds__(256) void lstm_cell_kernel(
        const float* __restrict__ gates, const float* __restrict__ bias,
        float* __restrict__ c_state, float* __restrict__ h_state,
        unsigned short* __restrict__ bf_dst1,   // may be null
        unsigned short* __restrict__ bf_dst2,   // may be null
        float* __restrict__ hlast)              // may be null
{
    int idx = blockIdx.x * 256 + threadIdx.x;   // 0 .. BB*HH-1
    int b   = idx >> 10;
    int col = idx & 1023;
    const float* gr = gates + (size_t)b * N4;

    float gi = gr[col]          + bias[col];
    float gf = gr[col + 1024]   + bias[col + 1024];
    float gg = gr[col + 2048]   + bias[col + 2048];
    float go = gr[col + 3072]   + bias[col + 3072];

    float i = fast_sigmoid(gi);
    float f = fast_sigmoid(gf);
    float g = fast_tanh(gg);
    float o = fast_sigmoid(go);

    float cy = i * g + f * c_state[idx];
    c_state[idx] = cy;
    float hy = o * fast_tanh(cy);
    h_state[idx] = hy;

    unsigned short hb = f32_to_bf16_rne(hy);
    if (bf_dst1) bf_dst1[(size_t)b * KK + col] = hb;
    if (bf_dst2) bf_dst2[(size_t)b * KK + col] = hb;
    if (hlast)   hlast[idx] = hy;
}

// ---------------------------------------------------------------------------
extern "C" void kernel_launch(void* const* d_in, const int* in_sizes, int n_in,
                              void* d_out, int out_size, void* d_ws, size_t ws_size,
                              hipStream_t stream) {
    (void)in_sizes; (void)n_in; (void)out_size; (void)ws_size;

    const float* x  = (const float*)d_in[0];   // [T,B,IN]
    const float* h0 = (const float*)d_in[1];   // [L,B,H]
    const float* c0 = (const float*)d_in[2];   // [L,B,H]
    const float* W0 = (const float*)d_in[3];   // [4H, IN+H]
    const float* b0 = (const float*)d_in[4];   // [4H]
    const float* W1 = (const float*)d_in[5];   // [4H, H+H]
    const float* b1 = (const float*)d_in[6];   // [4H]

    float* out    = (float*)d_out;
    float* hlast  = out;                                   // [T,B,H]
    float* hstate = out + (size_t)TT * BB * HH;            // [L,B,H] (live state)
    float* cstate = hstate + (size_t)LL * BB * HH;         // [L,B,H] (live state)

    // workspace layout
    unsigned short* Wt0 = (unsigned short*)d_ws;                   // 16 MB
    unsigned short* Wt1 = Wt0 + (size_t)N4 * KK;                   // 16 MB
    unsigned short* XH0 = Wt1 + (size_t)N4 * KK;                   // [64 x 2048] bf16
    unsigned short* XH1 = XH0 + (size_t)BB * KK;
    float*          gts = (float*)(XH1 + (size_t)BB * KK);         // [64 x 4096] f32

    const int WCONV_BLOCKS = (N4 * KK) / 256;
    const int EW_BLOCKS    = (BB * HH) / 256;

    // one-time-per-call prep: bf16 weight swizzle + state init
    wconvert_tiled_kernel<<<WCONV_BLOCKS, 256, 0, stream>>>(W0, Wt0);
    wconvert_tiled_kernel<<<WCONV_BLOCKS, 256, 0, stream>>>(W1, Wt1);
    hipMemcpyAsync(hstate, h0, sizeof(float) * LL * BB * HH,
                   hipMemcpyDeviceToDevice, stream);
    hipMemcpyAsync(cstate, c0, sizeof(float) * LL * BB * HH,
                   hipMemcpyDeviceToDevice, stream);
    pack_bf16_kernel<<<EW_BLOCKS, 256, 0, stream>>>(h0,                       XH0 + 1024);
    pack_bf16_kernel<<<EW_BLOCKS, 256, 0, stream>>>(h0 + (size_t)BB * HH,     XH1 + 1024);

    for (int t = 0; t < TT; ++t) {
        // layer 0
        pack_bf16_kernel<<<EW_BLOCKS, 256, 0, stream>>>(
            x + (size_t)t * BB * IN_D, XH0);
        gemm_gates_wmma_kernel<<<64, 256, 0, stream>>>(XH0, Wt0, gts);
        lstm_cell_kernel<<<EW_BLOCKS, 256, 0, stream>>>(
            gts, b0,
            cstate,                 // c state layer 0
            hstate,                 // h state layer 0
            XH1,                    // h0 -> layer-1 input half (this step)
            XH0 + 1024,             // h0 -> layer-0 recurrent half (next step)
            nullptr);

        // layer 1
        gemm_gates_wmma_kernel<<<64, 256, 0, stream>>>(XH1, Wt1, gts);
        lstm_cell_kernel<<<EW_BLOCKS, 256, 0, stream>>>(
            gts, b1,
            cstate + (size_t)BB * HH,
            hstate + (size_t)BB * HH,
            XH1 + 1024,             // h1 -> layer-1 recurrent half (next step)
            nullptr,
            hlast + (size_t)t * BB * HH);
    }
}
